// SVM_51462298140943
// MI455X (gfx1250) — matrix operand order repaired and checked by reading, not compiled, not run on hardware
//
#include <hip/hip_runtime.h>
#include <hip/hip_fp16.h>
#include <math.h>

typedef __attribute__((ext_vector_type(16))) _Float16 v16h;
typedef __attribute__((ext_vector_type(8)))  _Float16 v8h;
typedef __attribute__((ext_vector_type(8)))  float    v8f;

namespace {
constexpr int D_DIM      = 256;          // feature dim (fixed by reference)
constexpr int BM         = 64;           // x rows per block
constexpr int BN         = 32;           // data_set rows per tile
constexpr int NSPAN      = 512;          // N span per block (16 tiles)
constexpr int LDS_STRIDE = D_DIM + 8;    // halves; +16B pad -> 4-bank rotation
}

// --- pass 1a: fp32 x -> f16, row norms, zero accumulator --------------------
__global__ __launch_bounds__(256) void prep_x_kernel(
    const float* __restrict__ x, _Float16* __restrict__ xh,
    float* __restrict__ x2, float* __restrict__ acc)
{
    __shared__ float red[256];
    const int row = blockIdx.x;
    const int t   = threadIdx.x;
    float v = x[(size_t)row * D_DIM + t];
    xh[(size_t)row * D_DIM + t] = (_Float16)v;
    red[t] = v * v;
    __syncthreads();
    #pragma unroll
    for (int s = 128; s > 0; s >>= 1) {
        if (t < s) red[t] += red[t + s];
        __syncthreads();
    }
    if (t == 0) { x2[row] = red[0]; acc[row] = 0.0f; }
}

// --- pass 1b: fp32 data_set -> f16; swbuf[n] = (||s_n||^2, label*relu(lam)) -
__global__ __launch_bounds__(256) void prep_s_kernel(
    const float* __restrict__ s, const float* __restrict__ labels,
    const float* __restrict__ lambdas, _Float16* __restrict__ sh,
    float2* __restrict__ sw)
{
    __shared__ float red[256];
    const int row = blockIdx.x;
    const int t   = threadIdx.x;
    float v = s[(size_t)row * D_DIM + t];
    sh[(size_t)row * D_DIM + t] = (_Float16)v;
    red[t] = v * v;
    __syncthreads();
    #pragma unroll
    for (int st = 128; st > 0; st >>= 1) {
        if (t < st) red[t] += red[t + st];
        __syncthreads();
    }
    if (t == 0)
        sw[row] = make_float2(red[0], labels[row] * fmaxf(lambdas[row], 0.0f));
}

// --- pass 2: fused WMMA GEMM + RBF epilogue + deferred N-reduction ----------
__global__ __launch_bounds__(256) void svm_rbf_wmma_kernel(
    const _Float16* __restrict__ xh, const _Float16* __restrict__ sh,
    const float* __restrict__ x2, const float2* __restrict__ sw,
    float* __restrict__ acc, int N)
{
    // A staging buffer (33792 B) is dead after the register preload and is
    // exactly the size of TWO B tiles -> alias it for double buffering.
    __shared__ _Float16 ldsBuf[BM * LDS_STRIDE];     // == 2 * BN * LDS_STRIDE
    __shared__ float    ldsAcc[BM];
    __shared__ float    ldsX2[BM];
    _Float16* ldsA = ldsBuf;
    _Float16 (*ldsB)[BN * LDS_STRIDE] =
        reinterpret_cast<_Float16 (*)[BN * LDS_STRIDE]>(ldsBuf);

    const int t      = threadIdx.x;
    const int lane   = t & 31;
    const int wave   = t >> 5;       // 8 waves: 4 (M) x 2 (N)
    const int laneHi = lane >> 4;    // 0/1 half-wave
    const int lane15 = lane & 15;
    const int mi     = wave & 3;     // 16-row M subtile
    const int ni     = wave >> 2;    // 16-col N subtile

    const int mblock  = blockIdx.x * BM;
    const int nstart  = blockIdx.y * NSPAN;
    const int nend    = (nstart + NSPAN < N) ? (nstart + NSPAN) : N;
    const int nchunks = (nend - nstart + BN - 1) / BN;

    if (t < BM) { ldsAcc[t] = 0.0f; ldsX2[t] = x2[mblock + t]; }

    // stage A tile (64 rows x 256 halves) coalesced into LDS
    {
        const _Float16* src = xh + (size_t)mblock * D_DIM;
        #pragma unroll
        for (int i = 0; i < 8; ++i) {
            int c = t + i * 256;
            int row = c >> 5;
            int col = (c & 31) * 8;
            *(v8h*)(&ldsA[row * LDS_STRIDE + col]) =
                *(const v8h*)(src + row * D_DIM + col);
        }
    }
    __syncthreads();

    // preload this wave's A operands (ISA 7.12.2 wave32 layout) into 64 VGPRs;
    // they are invariant across all N chunks.
    v16h Areg[8];
    float x2r[8];
    {
        const _Float16* aBase =
            &ldsA[(mi * 16 + lane15) * LDS_STRIDE + laneHi * 8];
        #pragma unroll
        for (int kk = 0; kk < 8; ++kk) {
            v8h alo = *(const v8h*)(aBase + kk * 32);
            v8h ahi = *(const v8h*)(aBase + kk * 32 + 16);
            #pragma unroll
            for (int j = 0; j < 8; ++j) {
                Areg[kk][j] = alo[j]; Areg[kk][8 + j] = ahi[j];
            }
        }
        #pragma unroll
        for (int r = 0; r < 8; ++r)
            x2r[r] = -0.5f * ldsX2[mi * 16 + r + laneHi * 8];
    }
    __syncthreads();   // A staging consumed; buffer becomes B double-buffer

    v8h breg[4];
    auto load_b = [&](int n0) {
        #pragma unroll
        for (int i = 0; i < 4; ++i) {
            int c = t + i * 256;
            int row = c >> 5;
            int col = (c & 31) * 8;
            int n = n0 + row;
            if (n < N) {
                breg[i] = *(const v8h*)(sh + (size_t)n * D_DIM + col);
            } else {
                #pragma unroll
                for (int j = 0; j < 8; ++j) breg[i][j] = (_Float16)0.0f;
            }
        }
    };
    auto store_b = [&](int buf) {
        #pragma unroll
        for (int i = 0; i < 4; ++i) {
            int c = t + i * 256;
            int row = c >> 5;
            int col = (c & 31) * 8;
            *(v8h*)(&ldsB[buf][row * LDS_STRIDE + col]) = breg[i];
        }
    };

    load_b(nstart);
    store_b(0);
    __syncthreads();

    v8f psum;                      // deferred N-reduction accumulator
    #pragma unroll
    for (int j = 0; j < 8; ++j) psum[j] = 0.0f;

    int cur = 0;
    for (int chunk = 0; chunk < nchunks; ++chunk) {
        const int n0    = nstart + chunk * BN;
        const bool more = (chunk + 1) < nchunks;
        if (more) load_b(n0 + BN);                       // hoist next tile loads
        if ((chunk + 2) < nchunks) {                     // L2 prefetch 2 ahead
            int pn = n0 + 2 * BN + (t >> 3);
            if (pn < N)
                __builtin_prefetch(sh + (size_t)pn * D_DIM + (t & 7) * 32, 0, 3);
        }

        v8f cacc;
        #pragma unroll
        for (int j = 0; j < 8; ++j) cacc[j] = 0.0f;

        const _Float16* bBase =
            &ldsB[cur][(ni * 16 + lane15) * LDS_STRIDE + laneHi * 8];
        #pragma unroll
        for (int kk = 0; kk < 8; ++kk) {
            v8h blo = *(const v8h*)(bBase + kk * 32);
            v8h bhi = *(const v8h*)(bBase + kk * 32 + 16);
            v16h Bm;
            #pragma unroll
            for (int j = 0; j < 8; ++j) { Bm[j] = blo[j]; Bm[8 + j] = bhi[j]; }
            cacc = __builtin_amdgcn_wmma_f32_16x16x32_f16(
                false, Areg[kk], false, Bm, (short)0, cacc, false, false);
        }

        // RBF epilogue: psum[r] += exp(dot - 0.5||x||^2 - 0.5||s||^2) * w[n]
        {
            int n = n0 + ni * 16 + lane15;
            float es = 0.0f, wn = 0.0f;
            if (n < N) {
                float2 p = sw[n];
                es = -0.5f * p.x;
                wn = p.y;
            }
            #pragma unroll
            for (int r = 0; r < 8; ++r)
                psum[r] += __expf(cacc[r] + x2r[r] + es) * wn;
        }

        if (more) store_b(cur ^ 1);     // safe: readers retired at prev barrier
        __syncthreads();
        cur ^= 1;
    }

    // once-per-block cross-lane reduction over the 16-lane N groups
    #pragma unroll
    for (int r = 0; r < 8; ++r) {
        float v = psum[r];
        v += __shfl_xor(v, 1, 32);
        v += __shfl_xor(v, 2, 32);
        v += __shfl_xor(v, 4, 32);
        v += __shfl_xor(v, 8, 32);
        if (lane15 == 0)
            atomicAdd(&ldsAcc[mi * 16 + r + laneHi * 8], v);
    }
    __syncthreads();
    if (t < BM) atomicAdd(&acc[mblock + t], ldsAcc[t]);
}

// --- pass 3: out = tanh(acc + bias) -----------------------------------------
__global__ void finalize_kernel(const float* __restrict__ acc,
                                const float* __restrict__ bias,
                                float* __restrict__ out, int B)
{
    int i = blockIdx.x * blockDim.x + threadIdx.x;
    if (i < B) out[i] = tanhf(acc[i] + bias[0]);
}

extern "C" void kernel_launch(void* const* d_in, const int* in_sizes, int n_in,
                              void* d_out, int out_size, void* d_ws, size_t ws_size,
                              hipStream_t stream) {
    const float* x       = (const float*)d_in[0];
    const float* dset    = (const float*)d_in[1];
    const float* labels  = (const float*)d_in[2];
    const float* lambdas = (const float*)d_in[3];
    const float* bias    = (const float*)d_in[4];
    const int Brows = in_sizes[0] / D_DIM;   // 2048
    const int N     = in_sizes[2];           // 50000
    float* out = (float*)d_out;
    (void)n_in; (void)out_size; (void)ws_size;

    char* ws = (char*)d_ws;
    size_t off = 0;
    auto take = [&](size_t bytes) -> char* {
        char* p = ws + off;
        off = (off + bytes + 255) & ~(size_t)255;
        return p;
    };
    _Float16* xh   = (_Float16*)take((size_t)Brows * D_DIM * sizeof(_Float16));
    _Float16* sh   = (_Float16*)take((size_t)N * D_DIM * sizeof(_Float16));
    float*  x2buf  = (float*)take((size_t)Brows * sizeof(float));
    float2* swbuf  = (float2*)take((size_t)N * sizeof(float2));
    float*  accbuf = (float*)take((size_t)Brows * sizeof(float));

    prep_x_kernel<<<Brows, 256, 0, stream>>>(x, xh, x2buf, accbuf);
    prep_s_kernel<<<N, 256, 0, stream>>>(dset, labels, lambdas, sh, swbuf);

    dim3 grid(Brows / BM, (N + NSPAN - 1) / NSPAN);
    svm_rbf_wmma_kernel<<<grid, dim3(256), 0, stream>>>(
        xh, sh, x2buf, swbuf, accbuf, N);

    finalize_kernel<<<(Brows + 255) / 256, 256, 0, stream>>>(
        accbuf, bias, out, Brows);
}
